// MaskLRF_50044958933395
// MI455X (gfx1250) — matrix operand order repaired and checked by reading, not compiled, not run on hardware
//
#include <hip/hip_runtime.h>
#include <hip/hip_bf16.h>

// ---------------- problem constants ----------------
static constexpr int kB = 32;     // batches
static constexpr int kN = 8192;   // points per cloud
static constexpr int kG = 128;    // groups (FPS samples)
static constexpr int kS = 32;     // group size (KNN)
static constexpr int kENC = 384;  // output channels

typedef __attribute__((ext_vector_type(16))) _Float16 v16h;
typedef __attribute__((ext_vector_type(8)))  float    v8f;

union AV { v16h h; unsigned int u[8]; };

// =====================================================================
// Kernel 0: one-time weight conversion f32 -> f16 into workspace.
// =====================================================================
__global__ __launch_bounds__(256) void wcvt_kernel(
    const float* __restrict__ W2, const float* __restrict__ W3,
    const float* __restrict__ W4,
    _Float16* __restrict__ W2h, _Float16* __restrict__ W3h,
    _Float16* __restrict__ W4h) {
  const int i = blockIdx.x * 256 + threadIdx.x;
  if (i < 256 * 128) W2h[i] = (_Float16)W2[i];
  if (i < 512 * 512) W3h[i] = (_Float16)W3[i];
  if (i < 384 * 512) W4h[i] = (_Float16)W4[i];
}

// =====================================================================
// Kernel 1: farthest point sampling. One block per batch.
// =====================================================================
__global__ __launch_bounds__(512) void fps_kernel(const float* __restrict__ pc,
                                                  float* __restrict__ centers) {
  const int b = blockIdx.x;
  const int t = threadIdx.x;
  const float* p = pc + (size_t)b * kN * 3;

  __shared__ float d[kN];
  __shared__ float rv[512];
  __shared__ int   ri[512];

  const float INF = __builtin_inff();
  for (int i = t; i < kN; i += 512) d[i] = INF;
  __syncthreads();

  int last = 0;
  for (int g = 0; g < kG; ++g) {
    const float px = p[last * 3 + 0];
    const float py = p[last * 3 + 1];
    const float pz = p[last * 3 + 2];
    if (t == 0) {
      float* c = centers + ((size_t)b * kG + g) * 3;
      c[0] = px; c[1] = py; c[2] = pz;
    }
    float bv = -INF; int bi = 0;
    for (int i = t; i < kN; i += 512) {
      const float dx = p[i * 3 + 0] - px;
      const float dy = p[i * 3 + 1] - py;
      const float dz = p[i * 3 + 2] - pz;
      const float dist = dx * dx + dy * dy + dz * dz;
      const float dd = fminf(d[i], dist);
      d[i] = dd;
      if (dd > bv) { bv = dd; bi = i; }
    }
    rv[t] = bv; ri[t] = bi;
    __syncthreads();
    for (int off = 256; off > 0; off >>= 1) {
      if (t < off) {
        const float v2 = rv[t + off]; const int i2 = ri[t + off];
        if (v2 > rv[t] || (v2 == rv[t] && i2 < ri[t])) { rv[t] = v2; ri[t] = i2; }
      }
      __syncthreads();
    }
    last = ri[0];
    __syncthreads();
  }
}

// =====================================================================
// Kernel 2: KNN (32 smallest d2). One block per (b,g).
// =====================================================================
__global__ __launch_bounds__(256) void knn_kernel(const float* __restrict__ pc,
                                                  const float* __restrict__ centers,
                                                  int* __restrict__ nn) {
  const int bg = blockIdx.x;
  const int b  = bg / kG;
  const int t  = threadIdx.x;
  const float* p = pc + (size_t)b * kN * 3;
  const float* c = centers + (size_t)bg * 3;

  __shared__ float d2[kN];
  __shared__ float rv[256];
  __shared__ int   ri[256];

  const float INF = __builtin_inff();
  const float cx = c[0], cy = c[1], cz = c[2];
  const float qq = cx * cx + cy * cy + cz * cz;

  for (int i = t; i < kN; i += 256) {
    const float x = p[i * 3 + 0], y = p[i * 3 + 1], z = p[i * 3 + 2];
    const float tt = x * x + y * y + z * z;
    const float dt = cx * x + cy * y + cz * z;
    d2[i] = qq + tt - 2.0f * dt;
  }
  __syncthreads();

  for (int k = 0; k < kS; ++k) {
    float bv = INF; int bi = 0;
    for (int i = t; i < kN; i += 256) {
      const float v = d2[i];
      if (v < bv) { bv = v; bi = i; }
    }
    rv[t] = bv; ri[t] = bi;
    __syncthreads();
    for (int off = 128; off > 0; off >>= 1) {
      if (t < off) {
        const float v2 = rv[t + off]; const int i2 = ri[t + off];
        if (v2 < rv[t] || (v2 == rv[t] && i2 < ri[t])) { rv[t] = v2; ri[t] = i2; }
      }
      __syncthreads();
    }
    if (t == 0) {
      nn[(size_t)bg * kS + k] = ri[0];
      d2[ri[0]] = INF;
    }
    __syncthreads();
  }
}

// =====================================================================
// Kernel 3: LRF. One wave per group; lane = neighbor index.
// =====================================================================
__device__ __forceinline__ float wave_sum(float v) {
  for (int o = 16; o > 0; o >>= 1) v += __shfl_xor(v, o, 32);
  return v;
}
__device__ __forceinline__ float wave_max(float v) {
  for (int o = 16; o > 0; o >>= 1) v = fmaxf(v, __shfl_xor(v, o, 32));
  return v;
}

__device__ __forceinline__ void jacobi3(float A[3][3], float V[3][3]) {
  V[0][0] = 1.f; V[0][1] = 0.f; V[0][2] = 0.f;
  V[1][0] = 0.f; V[1][1] = 1.f; V[1][2] = 0.f;
  V[2][0] = 0.f; V[2][1] = 0.f; V[2][2] = 1.f;
  const int PP[3] = {0, 0, 1};
  const int QQ[3] = {1, 2, 2};
  for (int sweep = 0; sweep < 14; ++sweep) {
    for (int r = 0; r < 3; ++r) {
      const int p = PP[r], q = QQ[r];
      const float apq = A[p][q];
      if (fabsf(apq) > 1e-30f) {
        const float theta = (A[q][q] - A[p][p]) / (2.0f * apq);
        const float sgn = (theta >= 0.0f) ? 1.0f : -1.0f;
        const float tt = sgn / (fabsf(theta) + sqrtf(theta * theta + 1.0f));
        const float cth = 1.0f / sqrtf(tt * tt + 1.0f);
        const float sth = tt * cth;
        for (int k = 0; k < 3; ++k) {
          const float akp = A[k][p], akq = A[k][q];
          A[k][p] = cth * akp - sth * akq;
          A[k][q] = sth * akp + cth * akq;
        }
        for (int k = 0; k < 3; ++k) {
          const float apk = A[p][k], aqk = A[q][k];
          A[p][k] = cth * apk - sth * aqk;
          A[q][k] = sth * apk + cth * aqk;
        }
        for (int k = 0; k < 3; ++k) {
          const float vkp = V[k][p], vkq = V[k][q];
          V[k][p] = cth * vkp - sth * vkq;
          V[k][q] = sth * vkp + cth * vkq;
        }
      }
    }
  }
}

__device__ __forceinline__ void disamb(float& vx, float& vy, float& vz,
                                       float nx, float ny, float nz) {
  const float proj = vx * nx + vy * ny + vz * nz;
  float cnt = (proj > 0.0f) ? 1.0f : 0.0f;
  cnt = wave_sum(cnt);
  if (cnt < 16.0f) { vx = -vx; vy = -vy; vz = -vz; }  // n_pos < 0.5*S
}

__global__ __launch_bounds__(256) void lrf_kernel(const float* __restrict__ pc,
                                                  const float* __restrict__ centers,
                                                  const int* __restrict__ nn,
                                                  float* __restrict__ rot) {
  const int wave = threadIdx.x >> 5;
  const int lane = threadIdx.x & 31;
  const int bg = blockIdx.x * 8 + wave;
  const int b = bg / kG;

  const float* c = centers + (size_t)bg * 3;
  const int idx = nn[(size_t)bg * kS + lane];
  const float* pp = pc + ((size_t)b * kN + idx) * 3;

  const float nx = pp[0] - c[0];
  const float ny = pp[1] - c[1];
  const float nz = pp[2] - c[2];

  const float nrm = sqrtf(nx * nx + ny * ny + nz * nz);
  const float mx = wave_max(nrm);
  float w = mx - nrm;
  const float ws = wave_sum(w);
  w = w / (ws + 1e-6f);

  const float spx = 100.0f * nx, spy = 100.0f * ny, spz = 100.0f * nz;
  const float wx = w * spx, wy = w * spy, wz = w * spz;
  const float c00 = wave_sum(wx * spx);
  const float c01 = wave_sum(wx * spy);
  const float c02 = wave_sum(wx * spz);
  const float c11 = wave_sum(wy * spy);
  const float c12 = wave_sum(wy * spz);
  const float c22 = wave_sum(wz * spz);

  float A[3][3] = {{c00, c01, c02}, {c01, c11, c12}, {c02, c12, c22}};
  float V[3][3];
  jacobi3(A, V);

  float ev[3] = {A[0][0], A[1][1], A[2][2]};
  int i0 = 0, i1 = 1, i2 = 2, tmp;
  if (ev[i0] > ev[i1]) { tmp = i0; i0 = i1; i1 = tmp; }
  if (ev[i1] > ev[i2]) { tmp = i1; i1 = i2; i2 = tmp; }
  if (ev[i0] > ev[i1]) { tmp = i0; i0 = i1; i1 = tmp; }

  float zx = V[0][i0], zy = V[1][i0], zz = V[2][i0];
  float xx = V[0][i2], xy = V[1][i2], xz = V[2][i2];
  disamb(zx, zy, zz, nx, ny, nz);
  disamb(xx, xy, xz, nx, ny, nz);
  const float yx = zy * xz - zz * xy;
  const float yy = zz * xx - zx * xz;
  const float yz = zx * xy - zy * xx;

  float* rr = rot + ((size_t)bg * kS + lane) * 3;
  rr[0] = nx * zx + ny * zy + nz * zz;
  rr[1] = nx * yx + ny * yy + nz * yz;
  rr[2] = nx * xx + ny * xy + nz * xz;
}

// =====================================================================
// Kernel 4: encoder MLPs via WMMA. One block (8 waves) per group.
// f16 weights pre-converted; A fragments loaded once per (mtile, ks)
// and fed to 4-8 accumulators (register-blocked over N tiles).
// =====================================================================
__global__ __launch_bounds__(256) void encoder_kernel(
    const float* __restrict__ rot,
    const float* __restrict__ W1, const float* __restrict__ b1,
    const float* __restrict__ g1, const float* __restrict__ be1,
    const _Float16* __restrict__ W2h, const float* __restrict__ b2,
    const _Float16* __restrict__ W3h, const float* __restrict__ b3,
    const float* __restrict__ g2, const float* __restrict__ be2,
    const _Float16* __restrict__ W4h, const float* __restrict__ b4,
    float* __restrict__ out) {
  __shared__ _Float16 sf1[kS][128];   // 8 KB
  __shared__ _Float16 sf2[kS][256];   // 16 KB
  __shared__ _Float16 sfg[256];       // 0.5 KB
  __shared__ _Float16 sf3[kS][512];   // 32 KB
  __shared__ float    pm[4][kENC];    // 6 KB  (stage-4 partial maxes)

  const int bg = blockIdx.x;
  const int t = threadIdx.x;
  const int lane = t & 31;
  const int wave = t >> 5;
  const int lrow = lane & 15;
  const int lhi = lane >> 4;
  const int mt   = wave >> 2;   // mtile owned by this wave (0 or 1)
  const int ncol = wave & 3;    // base N-tile column class
  const float bns = 1.0f / sqrtf(1.0f + 1e-5f);
  const float NEGINF = -__builtin_inff();

  const float* r = rot + (size_t)bg * kS * 3;
  const v8f vzero = {};

  // ---- stage 1: f1 = relu(bn(rot @ W1^T + b1)) [32 x 128], f32 VALU ----
  for (int e = t; e < kS * 128; e += 256) {
    const int s = e >> 7, o = e & 127;
    float x = r[s * 3 + 0] * W1[o * 3 + 0] +
              r[s * 3 + 1] * W1[o * 3 + 1] +
              r[s * 3 + 2] * W1[o * 3 + 2] + b1[o];
    x = x * (g1[o] * bns) + be1[o];
    sf1[s][o] = (_Float16)fmaxf(x, 0.0f);
  }
  __syncthreads();

  // ---- stage 2: f2 = f1 @ W2^T + b2   [32 x 256], K=128 ----
  {
    v8f acc[4];
#pragma unroll
    for (int i = 0; i < 4; ++i) acc[i] = vzero;
    const int m = mt * 16 + lrow;
    for (int ks = 0; ks < 4; ++ks) {
      AV av;
#pragma unroll
      for (int j = 0; j < 8; ++j) {   // contiguous 16-bit A layout (k = 16h + s), pairs as b32
        const int kb = 2 * j + (lhi ? 16 : 0) + ks * 32;
        av.u[j] = *(const unsigned int*)&sf1[m][kb];
      }
      const int kofs = ks * 32 + (lhi ? 16 : 0);
#pragma unroll
      for (int i = 0; i < 4; ++i) {
        const int n = (ncol + 4 * i) * 16 + lrow;
        const v16h bf = *(const v16h*)(W2h + (size_t)n * 128 + kofs);
        acc[i] = __builtin_amdgcn_wmma_f32_16x16x32_f16(false, av.h, false, bf,
                                                        (short)0, acc[i], false, false);
      }
    }
    const int mrow = mt * 16 + lhi * 8;
#pragma unroll
    for (int i = 0; i < 4; ++i) {
      const int n = (ncol + 4 * i) * 16 + lrow;
      const float bb = b2[n];
#pragma unroll
      for (int rr = 0; rr < 8; ++rr)
        sf2[mrow + rr][n] = (_Float16)(acc[i][rr] + bb);
    }
  }
  __syncthreads();

  // ---- fg = max_s f2 ----
  for (int p = t; p < 256; p += 256) {
    float mxv = NEGINF;
    for (int s = 0; s < kS; ++s) mxv = fmaxf(mxv, (float)sf2[s][p]);
    sfg[p] = (_Float16)mxv;
  }
  __syncthreads();

  // ---- stage 3: f3 = relu(bn([fg | f2] @ W3^T + b3)) [32 x 512], K=512 ----
  {
    v8f acc[8];
#pragma unroll
    for (int i = 0; i < 8; ++i) acc[i] = vzero;
    const int m = mt * 16 + lrow;
    for (int ks = 0; ks < 16; ++ks) {
      AV av;
#pragma unroll
      for (int j = 0; j < 8; ++j) {
        const int kb = 2 * j + (lhi ? 16 : 0) + ks * 32;
        if (kb < 256) av.u[j] = *(const unsigned int*)&sfg[kb];
        else          av.u[j] = *(const unsigned int*)&sf2[m][kb - 256];
      }
      const int kofs = ks * 32 + (lhi ? 16 : 0);
#pragma unroll
      for (int i = 0; i < 8; ++i) {
        const int n = (ncol + 4 * i) * 16 + lrow;
        const v16h bf = *(const v16h*)(W3h + (size_t)n * 512 + kofs);
        acc[i] = __builtin_amdgcn_wmma_f32_16x16x32_f16(false, av.h, false, bf,
                                                        (short)0, acc[i], false, false);
      }
    }
    const int mrow = mt * 16 + lhi * 8;
#pragma unroll
    for (int i = 0; i < 8; ++i) {
      const int n = (ncol + 4 * i) * 16 + lrow;
      const float scale = g2[n] * bns;
      const float bb = b3[n];
      const float sh = be2[n];
#pragma unroll
      for (int rr = 0; rr < 8; ++rr) {
        const float x = (acc[i][rr] + bb) * scale + sh;
        sf3[mrow + rr][n] = (_Float16)fmaxf(x, 0.0f);
      }
    }
  }
  __syncthreads();

  // ---- stage 4: out = max_s(f3 @ W4^T + b4)  [384], K=512, fused max ----
  {
    v8f acc[6];
#pragma unroll
    for (int i = 0; i < 6; ++i) acc[i] = vzero;
    const int m = mt * 16 + lrow;
    for (int ks = 0; ks < 16; ++ks) {
      AV av;
#pragma unroll
      for (int j = 0; j < 8; ++j) {
        const int kb = 2 * j + (lhi ? 16 : 0) + ks * 32;
        av.u[j] = *(const unsigned int*)&sf3[m][kb];
      }
      const int kofs = ks * 32 + (lhi ? 16 : 0);
#pragma unroll
      for (int i = 0; i < 6; ++i) {
        const int n = (ncol + 4 * i) * 16 + lrow;
        const v16h bf = *(const v16h*)(W4h + (size_t)n * 512 + kofs);
        acc[i] = __builtin_amdgcn_wmma_f32_16x16x32_f16(false, av.h, false, bf,
                                                        (short)0, acc[i], false, false);
      }
    }
    // lane covers 8 of 32 rows per tile; writer class = (lane-half, mtile)
    const int slot = lhi * 2 + mt;
#pragma unroll
    for (int i = 0; i < 6; ++i) {
      float mv = NEGINF;
#pragma unroll
      for (int rr = 0; rr < 8; ++rr) mv = fmaxf(mv, acc[i][rr]);
      pm[slot][(ncol + 4 * i) * 16 + lrow] = mv;
    }
  }
  __syncthreads();
  for (int n = t; n < kENC; n += 256) {
    const float v = fmaxf(fmaxf(pm[0][n], pm[1][n]), fmaxf(pm[2][n], pm[3][n]));
    out[(size_t)bg * kENC + n] = v + b4[n];
  }
}

// =====================================================================
// Host launcher
// =====================================================================
extern "C" void kernel_launch(void* const* d_in, const int* in_sizes, int n_in,
                              void* d_out, int out_size, void* d_ws, size_t ws_size,
                              hipStream_t stream) {
  (void)in_sizes; (void)n_in; (void)out_size; (void)ws_size;

  const float* pc  = (const float*)d_in[0];
  const float* W1  = (const float*)d_in[1];
  const float* b1  = (const float*)d_in[2];
  const float* g1  = (const float*)d_in[3];
  const float* be1 = (const float*)d_in[4];
  const float* W2  = (const float*)d_in[5];
  const float* b2  = (const float*)d_in[6];
  const float* W3  = (const float*)d_in[7];
  const float* b3  = (const float*)d_in[8];
  const float* g2  = (const float*)d_in[9];
  const float* be2 = (const float*)d_in[10];
  const float* W4  = (const float*)d_in[11];
  const float* b4  = (const float*)d_in[12];
  float* out = (float*)d_out;

  // workspace layout (all offsets 32B aligned)
  float* centers = (float*)d_ws;                            // B*G*3 floats
  int*   nn      = (int*)(centers + (size_t)kB * kG * 3);   // B*G*S ints
  float* rotbuf  = (float*)(nn + (size_t)kB * kG * kS);     // B*G*S*3 floats
  _Float16* W2h  = (_Float16*)(rotbuf + (size_t)kB * kG * kS * 3);
  _Float16* W3h  = W2h + (size_t)256 * 128;
  _Float16* W4h  = W3h + (size_t)512 * 512;

  wcvt_kernel<<<(512 * 512 + 255) / 256, 256, 0, stream>>>(W2, W3, W4, W2h, W3h, W4h);
  fps_kernel<<<kB, 512, 0, stream>>>(pc, centers);
  knn_kernel<<<kB * kG, 256, 0, stream>>>(pc, centers, nn);
  lrf_kernel<<<(kB * kG) / 8, 256, 0, stream>>>(pc, centers, nn, rotbuf);
  encoder_kernel<<<kB * kG, 256, 0, stream>>>(rotbuf, W1, b1, g1, be1, W2h, b2,
                                              W3h, b3, g2, be2, W4h, b4, out);
}
